// Lfilter_53420803228289
// MI455X (gfx1250) — matrix operand order repaired and checked by reading, not compiled, not run on hardware
//
#include <hip/hip_runtime.h>
#include <hip/hip_bf16.h>

typedef __attribute__((ext_vector_type(16))) _Float16 v16h;
typedef __attribute__((ext_vector_type(8)))  _Float16 v8h;
typedef __attribute__((ext_vector_type(8)))  float    v8f;

#define NB 4      // batch
#define NF 128    // feature / hidden
#define NT 32000  // time
#define NG 384    // 3*hidden (r,z,n)
#define NH 64     // filter bank size
#define NK 32     // FIR taps
#define PFD 12    // GRU X-prefetch distance (steps)

__device__ __forceinline__ float sigmoid_f(float x) {
  return 1.f / (1.f + __expf(-x));
}
__device__ __forceinline__ float tanh_f(float x) {
  float e = __expf(-2.f * x);
  return (1.f - e) / (1.f + e);
}

// ---------------------------------------------------------------------------
// Prep: collapse filter bank -> keff[32]; convert GRU weights to f16.
// Wih16 / Whh16 layout: [dir][g][f]  (row-major, 128 cols)
// ---------------------------------------------------------------------------
__global__ void prep_kernel(const float* __restrict__ kern,
                            const float* __restrict__ kw,
                            const float* __restrict__ Wihf,
                            const float* __restrict__ Wihb,
                            const float* __restrict__ Whhf,
                            const float* __restrict__ Whhb,
                            float* __restrict__ keff,
                            _Float16* __restrict__ Wih16,
                            _Float16* __restrict__ Whh16) {
  int idx = blockIdx.x * 256 + threadIdx.x;
  const int n = NG * NF;  // 49152
  if (idx < n) {
    Wih16[idx]     = (_Float16)Wihf[idx];
    Wih16[n + idx] = (_Float16)Wihb[idx];
    Whh16[idx]     = (_Float16)Whhf[idx];
    Whh16[n + idx] = (_Float16)Whhb[idx];
  }
  if (blockIdx.x == 0 && threadIdx.x < NK) {
    float s = 0.f;
    for (int h = 0; h < NH; ++h)
      s += kern[h * NK + threadIdx.x] * kw[h * NK + threadIdx.x];
    keff[threadIdx.x] = s;
  }
}

// ---------------------------------------------------------------------------
// FIR (causal, 32 taps) + PReLU, written as f16 in [b*T+t][f] layout (GEMM A).
// grid: (NT/2, NB), block: 256 (lane->f contiguous stores)
// ---------------------------------------------------------------------------
__global__ __launch_bounds__(256) void fir_kernel(const float* __restrict__ x,
                                                  const float* __restrict__ keff,
                                                  const float* __restrict__ prelu_a,
                                                  _Float16* __restrict__ seq16) {
  __shared__ float kf[NK];
  int tid = threadIdx.x;
  if (tid < NK) kf[tid] = keff[tid];
  __syncthreads();
  int f = tid & (NF - 1);
  int t = blockIdx.x * 2 + (tid >> 7);
  int b = blockIdx.y;
  const float* xr = x + ((size_t)b * NF + f) * NT;
  float a = prelu_a[0];
  float acc = 0.f;
  int kmax = (t + 1 < NK) ? (t + 1) : NK;
  for (int k = 0; k < kmax; ++k) acc += kf[k] * xr[t - k];
  float o = (acc >= 0.f) ? acc : a * acc;
  seq16[((size_t)b * NT + t) * NF + f] = (_Float16)o;
}

// ---------------------------------------------------------------------------
// Input projection GEMM: X[dir][t][b][g] = seq[m][:] . Wih[dir][g][:] + bih[g]
// M = B*T = 128000 (8000 tiles of 16), N = 384, K = 128 (4 chunks).
// grid: (8000, 2), block: 256. Wave w computes n-tiles {w, w+8, w+16}
// (one per gate) reusing its A tile 3x. Output layout is step-contiguous
// so the recurrence reads one 3KB block per step.
// ---------------------------------------------------------------------------
__global__ __launch_bounds__(256) void xproj_kernel(
    const _Float16* __restrict__ seq16,
    const _Float16* __restrict__ Wih16,
    const float* __restrict__ bih_f,
    const float* __restrict__ bih_b,
    _Float16* __restrict__ X) {
  int tid  = threadIdx.x;
  int wv   = tid >> 5;
  int lane = tid & 31;
  int col  = lane & 15;
  int hi   = lane >> 4;  // 0: lanes 0-15, 1: lanes 16-31
  int mt   = blockIdx.x;  // 0..7999
  int dir  = blockIdx.y;

  const _Float16* Wd  = Wih16 + (size_t)dir * NG * NF;
  const float*    bih = (dir == 0) ? bih_f : bih_b;
  _Float16* Xd = X + (size_t)dir * ((size_t)NB * NT) * NG;

  // A tiles: lane holds row (mt*16+col); contiguous 16 K values (hi selects half)
  int row = mt * 16 + col;
  v16h A[4];
#pragma unroll
  for (int kc = 0; kc < 4; ++kc) {
    const _Float16* src = seq16 + (size_t)row * NF + kc * 32 + hi * 16;
    v8h lo = *(const v8h*)src;
    v8h hv = *(const v8h*)(src + 8);
    v16h a;
#pragma unroll
    for (int i = 0; i < 8; ++i) { a[i] = lo[i]; a[i + 8] = hv[i]; }
    A[kc] = a;
  }

#pragma unroll
  for (int jj = 0; jj < 3; ++jj) {
    int nt = wv + jj * 8;     // gate jj, 16-col slice wv
    int g  = nt * 16 + col;
    // B tile: lane holds column g, contiguous 16 K values (hi selects half)
    v16h Bv[4];
#pragma unroll
    for (int kc = 0; kc < 4; ++kc)
      Bv[kc] = *(const v16h*)(Wd + (size_t)g * NF + kc * 32 + hi * 16);

    v8f acc = {0.f, 0.f, 0.f, 0.f, 0.f, 0.f, 0.f, 0.f};
#pragma unroll
    for (int kc = 0; kc < 4; ++kc)
      acc = __builtin_amdgcn_wmma_f32_16x16x32_f16(false, A[kc], false, Bv[kc],
                                                   (short)0, acc, false, false);

    float bb = bih[g];
#pragma unroll
    for (int v = 0; v < 8; ++v) {
      int r = mt * 16 + v + hi * 8;       // global m index (tiles never straddle b)
      int b = r / NT;
      int t = r - b * NT;
      Xd[((size_t)t * NB + b) * NG + g] = (_Float16)(acc[v] + bb);
    }
  }
}

// ---------------------------------------------------------------------------
// GRU recurrence. grid = 2 (dir), block = 256 (8 waves).
// Wave w owns hidden columns [16w, 16w+16): 12 WMMAs/step (3 gates x 4
// K-chunks, interleaved kc-outer for pipelining). h double-buffered in LDS
// as f16. X block for step t+PFD is prefetched each step (global_prefetch).
// ---------------------------------------------------------------------------
__global__ __launch_bounds__(256) void gru_kernel(
    const _Float16* __restrict__ X,       // [2][T][B][384]
    const _Float16* __restrict__ Whh16,   // [2][384][128]
    const float* __restrict__ bhh_f,
    const float* __restrict__ bhh_b,
    float* __restrict__ hout) {           // [2][B][T][128]
  const int dir  = blockIdx.x;
  const int tid  = threadIdx.x;
  const int w    = tid >> 5;
  const int lane = tid & 31;
  const int col  = lane & 15;
  const int hi   = lane >> 4;

  __shared__ __align__(32) _Float16 hbuf[2][16][NF];
  for (int i = tid; i < 2 * 16 * NF; i += 256)
    ((_Float16*)hbuf)[i] = (_Float16)0.f;

  const _Float16* Xd  = X + (size_t)dir * ((size_t)NB * NT) * NG;
  const _Float16* Wd  = Whh16 + (size_t)dir * NG * NF;
  const float*    bhh = (dir == 0) ? bhh_f : bhh_b;
  float* out = hout + (size_t)dir * ((size_t)NB * NT) * NF;

  // Preload recurrent weights (B matrices) into registers: 3 gates x 4 chunks
  v16h Bm[3][4];
#pragma unroll
  for (int j = 0; j < 3; ++j) {
    int g = j * NF + w * 16 + col;
#pragma unroll
    for (int kc = 0; kc < 4; ++kc)
      Bm[j][kc] = *(const v16h*)(Wd + (size_t)g * NF + kc * 32 + hi * 16);
  }
  float bias[3];
#pragma unroll
  for (int j = 0; j < 3; ++j) bias[j] = bhh[j * NF + w * 16 + col];

  float hreg[NB] = {0.f, 0.f, 0.f, 0.f};  // h_old for this wave's column
  __syncthreads();

  int p = 0;
  for (int t = 0; t < NT; ++t) {
    int tt = (dir == 0) ? t : (NT - 1 - t);

    // Prefetch the X block PFD steps ahead (24 x 128B lines; wave 0 only)
    {
      int tp = t + PFD;
      if (tp > NT - 1) tp = NT - 1;
      int ttp = (dir == 0) ? tp : (NT - 1 - tp);
      const char* pf = (const char*)(Xd + (size_t)ttp * (NB * NG));
      if (tid < 24) __builtin_prefetch(pf + tid * 128, 0, 3);
    }

    // A tile from LDS h (rows 0..3 live, 4..15 stay zero)
    v16h A[4];
#pragma unroll
    for (int kc = 0; kc < 4; ++kc) {
      const _Float16* src = &hbuf[p][col][kc * 32 + hi * 16];
      v8h lo = *(const v8h*)src;
      v8h hv = *(const v8h*)(src + 8);
      v16h a;
#pragma unroll
      for (int i = 0; i < 8; ++i) { a[i] = lo[i]; a[i + 8] = hv[i]; }
      A[kc] = a;
    }

    const v8f vzero = {0.f, 0.f, 0.f, 0.f, 0.f, 0.f, 0.f, 0.f};
    v8f acc[3] = {vzero, vzero, vzero};
#pragma unroll
    for (int kc = 0; kc < 4; ++kc) {  // kc-outer: 3 independent chains interleave
#pragma unroll
      for (int j = 0; j < 3; ++j)
        acc[j] = __builtin_amdgcn_wmma_f32_16x16x32_f16(
            false, A[kc], false, Bm[j][kc], (short)0, acc[j], false, false);
    }

    // Precomputed input projections (already include b_ih); step-contiguous
    float xg[3][NB];
#pragma unroll
    for (int v = 0; v < NB; ++v) {
      const _Float16* xp = Xd + ((size_t)tt * NB + v) * NG + w * 16 + col;
      xg[0][v] = (float)xp[0];
      xg[1][v] = (float)xp[NF];
      xg[2][v] = (float)xp[2 * NF];
    }

    float hnew[NB];
#pragma unroll
    for (int v = 0; v < NB; ++v) {
      float r = sigmoid_f(xg[0][v] + acc[0][v] + bias[0]);
      float z = sigmoid_f(xg[1][v] + acc[1][v] + bias[1]);
      float nn = tanh_f(xg[2][v] + r * (acc[2][v] + bias[2]));
      hnew[v] = (1.f - z) * nn + z * hreg[v];
      hreg[v] = hnew[v];
    }

    if (lane < 16) {  // divergence is after the WMMAs; barrier reconverges
#pragma unroll
      for (int v = 0; v < NB; ++v) {
        hbuf[1 - p][v][w * 16 + col] = (_Float16)hnew[v];
        out[((size_t)v * NT + tt) * NF + w * 16 + col] = hnew[v];
      }
    }
    p ^= 1;
    __syncthreads();
  }
}

// ---------------------------------------------------------------------------
// out[b][f][t] = hf[b][t][f] + hb[b][t][f]  -- LDS-tiled 32x32 transpose.
// grid: (NT/32, NF/32, NB), block: 256
// ---------------------------------------------------------------------------
__global__ __launch_bounds__(256) void combine_kernel(
    const float* __restrict__ hf, const float* __restrict__ hb,
    float* __restrict__ out) {
  __shared__ float tile[32][33];
  int t0 = blockIdx.x * 32;
  int f0 = blockIdx.y * 32;
  int b  = blockIdx.z;
  int tid = threadIdx.x;
#pragma unroll
  for (int i = tid; i < 1024; i += 256) {
    int f = i & 31, t = i >> 5;  // lanes -> f: coalesced 128B reads
    size_t src = ((size_t)b * NT + (t0 + t)) * NF + (f0 + f);
    tile[t][f] = hf[src] + hb[src];
  }
  __syncthreads();
#pragma unroll
  for (int i = tid; i < 1024; i += 256) {
    int t = i & 31, f = i >> 5;  // lanes -> t: coalesced 128B writes
    out[((size_t)b * NF + (f0 + f)) * NT + (t0 + t)] = tile[t][f];
  }
}

// ---------------------------------------------------------------------------
extern "C" void kernel_launch(void* const* d_in, const int* in_sizes, int n_in,
                              void* d_out, int out_size, void* d_ws, size_t ws_size,
                              hipStream_t stream) {
  const float* x     = (const float*)d_in[0];
  const float* kern  = (const float*)d_in[1];
  const float* kw    = (const float*)d_in[2];
  const float* pre_a = (const float*)d_in[3];
  const float* Wihf  = (const float*)d_in[4];
  const float* Whhf  = (const float*)d_in[5];
  const float* bihf  = (const float*)d_in[6];
  const float* bhhf  = (const float*)d_in[7];
  const float* Wihb  = (const float*)d_in[8];
  const float* Whhb  = (const float*)d_in[9];
  const float* bihb  = (const float*)d_in[10];
  const float* bhhb  = (const float*)d_in[11];
  float* out = (float*)d_out;

  char* ws = (char*)d_ws;
  // 256-byte aligned sub-buffers
  float*    keff  = (float*)(ws + 0);                       // 32 f32
  _Float16* Wih16 = (_Float16*)(ws + 256);                  // 2*384*128 f16
  _Float16* Whh16 = (_Float16*)(ws + 256 + 196608);         // 2*384*128 f16
  _Float16* seq16 = (_Float16*)(ws + 393472);               // 128000*128 f16
  _Float16* X     = (_Float16*)(ws + 33161472ull);          // 2*128000*384 f16
  float*    hbufs = (float*)(ws + 229769472ull);            // 2*128000*128 f32

  prep_kernel<<<192, 256, 0, stream>>>(kern, kw, Wihf, Wihb, Whhf, Whhb,
                                       keff, Wih16, Whh16);
  fir_kernel<<<dim3(NT / 2, NB), 256, 0, stream>>>(x, keff, pre_a, seq16);
  xproj_kernel<<<dim3((NB * NT) / 16, 2), 256, 0, stream>>>(seq16, Wih16,
                                                            bihf, bihb, X);
  gru_kernel<<<2, 256, 0, stream>>>(X, Whh16, bhhf, bhhb, hbufs);
  combine_kernel<<<dim3(NT / 32, NF / 32, NB), 256, 0, stream>>>(
      hbufs, hbufs + (size_t)NB * NT * NF, out);
}
